// HybridMemoryMultiFocalPercent_65403761983943
// MI455X (gfx1250) — matrix operand order repaired and checked by reading, not compiled, not run on hardware
//
#include <hip/hip_runtime.h>
#include <hip/hip_bf16.h>

// GEMM: out[b, m] = sum_k inputs[b, k] * features[m, k]
//   inputs:   [512, 256]  f32
//   features: [65536, 256] f32
//   out:      [512, 65536] f32
//
// Strategy: split-bf16 (hi/lo) x3 WMMA products on v_wmma_f32_16x16x32_bf16
// for ~f32 accuracy at bf16 matrix-core rate. Memory-bound (~192MB @ 23.3TB/s).

typedef __attribute__((ext_vector_type(16))) __bf16 v16bf;
typedef __attribute__((ext_vector_type(8)))  float  v8f;
typedef __attribute__((ext_vector_type(4)))  float  f4;

#define NUM_MEMORY   65536
#define NUM_FEATURES 256
#define BATCH        512

// Load 16 f32 (two contiguous runs of 8) and split each into bf16 hi + bf16 lo
// residual, packed into v16bf fragments (element 2j,2j+1 live in VGPR j).
__device__ __forceinline__ void load_split16(const float* __restrict__ p0,
                                             const float* __restrict__ p1,
                                             v16bf& hi, v16bf& lo) {
  const f4* q0 = (const f4*)p0;
  const f4* q1 = (const f4*)p1;
  f4 x[4];
  x[0] = q0[0]; x[1] = q0[1];   // 8 floats, run 0
  x[2] = q1[0]; x[3] = q1[1];   // 8 floats, run 1
#pragma unroll
  for (int i = 0; i < 16; ++i) {
    float f  = x[i >> 2][i & 3];
    __bf16 h = (__bf16)f;
    hi[i] = h;
    lo[i] = (__bf16)(f - (float)h);
  }
}

__global__ __launch_bounds__(128)
void hm_logits_wmma(const float* __restrict__ inputs,
                    const float* __restrict__ features,
                    float* __restrict__ out) {
  const int lane = threadIdx.x & 31;
  const int wave = threadIdx.x >> 5;   // 0..3, each wave owns a 64-wide n strip
  const int l    = lane & 15;
  const int half = lane >> 4;          // lane-half per ISA fragment layout

  const int bBase = blockIdx.y * 32;              // 2 b-tiles of 16
  const int nBase = blockIdx.x * 256 + wave * 64; // 4 n-tiles of 16

  v8f acc[2][4] = {};                  // f32 16x16 accumulators

  for (int kb = 0; kb < NUM_FEATURES; kb += 32) {
    // ---- A fragments: inputs rows bBase.. (16x32 bf16 layout) ----
    // lane-half 0 holds K = kb+0..7 (v0-3) and kb+16..23 (v4-7);
    // lane-half 1 holds K = kb+8..15 and kb+24..31.
    v16bf ahi[2], alo[2];
#pragma unroll
    for (int bt = 0; bt < 2; ++bt) {
      const float* ar =
          inputs + (size_t)(bBase + bt * 16 + l) * NUM_FEATURES + kb + half * 8;
      load_split16(ar, ar + 16, ahi[bt], alo[bt]);
    }

    // ---- B fragments per n-tile: B[k,n] = features[n,k] (32x16 bf16) ----
    // lane = column n; lane-half 0 holds K = kb+0..7 and kb+16..23,
    // half 1 holds kb+8..15 and kb+24..31 (same K order as A).
#pragma unroll
    for (int nt = 0; nt < 4; ++nt) {
      const float* br =
          features + (size_t)(nBase + nt * 16 + l) * NUM_FEATURES + kb + half * 8;
      if (kb + 32 < NUM_FEATURES)
        __builtin_prefetch(br + 32, 0, 0);   // global_prefetch next K chunk
      v16bf bhi, blo;
      load_split16(br, br + 16, bhi, blo);

#pragma unroll
      for (int bt = 0; bt < 2; ++bt) {
        // 3-product split-bf16: hi*hi + hi*lo + lo*hi  (~f32 accuracy)
        acc[bt][nt] = __builtin_amdgcn_wmma_f32_16x16x32_bf16(
            false, ahi[bt], false, bhi, (short)0, acc[bt][nt], false, false);
        acc[bt][nt] = __builtin_amdgcn_wmma_f32_16x16x32_bf16(
            false, ahi[bt], false, blo, (short)0, acc[bt][nt], false, false);
        acc[bt][nt] = __builtin_amdgcn_wmma_f32_16x16x32_bf16(
            false, alo[bt], false, bhi, (short)0, acc[bt][nt], false, false);
      }
    }
  }

  // ---- Store: C/D layout — VGPR r holds M=r (lanes 0-15), M=r+8 (lanes 16-31)
#pragma unroll
  for (int bt = 0; bt < 2; ++bt) {
#pragma unroll
    for (int nt = 0; nt < 4; ++nt) {
#pragma unroll
      for (int r = 0; r < 8; ++r) {
        const int b = bBase + bt * 16 + half * 8 + r;
        const int n = nBase + nt * 16 + l;
        out[(size_t)b * NUM_MEMORY + n] = acc[bt][nt][r];
      }
    }
  }
}

extern "C" void kernel_launch(void* const* d_in, const int* in_sizes, int n_in,
                              void* d_out, int out_size, void* d_ws, size_t ws_size,
                              hipStream_t stream) {
  (void)in_sizes; (void)n_in; (void)out_size; (void)d_ws; (void)ws_size;
  // setup_inputs order: inputs, indexes, features, mIoU, IoU
  const float* inputs   = (const float*)d_in[0];
  const float* features = (const float*)d_in[2];
  float* out            = (float*)d_out;

  dim3 grid(NUM_MEMORY / 256, BATCH / 32);  // (n blocks, b blocks)
  dim3 block(128);                           // 4 wave32
  hipLaunchKernelGGL(hm_logits_wmma, grid, block, 0, stream,
                     inputs, features, out);
}
